// Block_2430951489990
// MI455X (gfx1250) — matrix operand order repaired and checked by reading, not compiled, not run on hardware
//
#include <hip/hip_runtime.h>

typedef unsigned int u32;
typedef unsigned short u16;
typedef unsigned long long u64;
typedef __attribute__((ext_vector_type(16))) __bf16 v16bf;
typedef __attribute__((ext_vector_type(8)))  float  v8f;
typedef __attribute__((ext_vector_type(4)))  u32    v4u;

// ---------------- helpers ----------------

__device__ __forceinline__ u16 f2bf(float f) {
  u32 u = __builtin_bit_cast(u32, f);
  u32 r = u + 0x7FFFu + ((u >> 16) & 1u);   // round-to-nearest-even
  return (u16)(r >> 16);
}
__device__ __forceinline__ u32 pk2bf(float lo, float hi) {
  return (u32)f2bf(lo) | ((u32)f2bf(hi) << 16);
}

struct FragBF { union { u32 u[8]; v4u q[2]; v16bf v; }; };

__device__ __forceinline__ v8f wmma_bf16(const FragBF& a, const FragBF& b, v8f c) {
  return __builtin_amdgcn_wmma_f32_16x16x32_bf16(false, a.v, false, b.v,
                                                 (short)0, c, false, false);
}

// CDNA5 async copy: 16B per lane, global -> LDS, ASYNCcnt-tracked.
__device__ __forceinline__ void async_b128(u32 lds_off, u32 voff, u64 base) {
  asm volatile("global_load_async_to_lds_b128 %0, %1, %2"
               :: "v"(lds_off), "v"(voff), "s"(base) : "memory");
}
__device__ __forceinline__ u32 lds_addr(const void* p) {
  return (u32)(uintptr_t)p;   // flat->LDS aperture keeps DS offset in addr[31:0]
}

// A-matrix 16x32 bf16 fragment from row-major LDS (stride u16; rows 16B-aligned).
// lanes0-15 M=lane, K {0..15}; lanes16-31 K {16..31}.
__device__ __forceinline__ void load_a_frag(FragBF& f, const u16* lds, int stride, int lane) {
  int m  = lane & 15;
  int kd = (lane < 16) ? 0 : 8;
  const v4u* row = (const v4u*)((const u32*)(lds + m * stride) + kd);
  f.q[0] = row[0];          // dwords kd..kd+3
  f.q[1] = row[1];          // dwords kd+4..kd+7
}

// B = Ktile^T fragment from row-major Ktile LDS: 8 contiguous dwords per lane.
__device__ __forceinline__ void load_b_frag_rm(FragBF& f, const u16* lds, int stride, int lane) {
  const u32* row = (const u32*)(lds + (lane & 15) * stride) + ((lane < 16) ? 0 : 8);
  f.q[0] = ((const v4u*)row)[0];
  f.q[1] = ((const v4u*)row)[1];
}

// B fragment from column-major pair-packed LDS tile [n][k2] (stride dwords, padded):
// lane n reads k2 = base..base+7 contiguous dwords.
__device__ __forceinline__ void load_b_frag_t(FragBF& f, const u32* lds, int stride, int lane) {
  const u32* p = lds + (lane & 15) * stride + ((lane < 16) ? 0 : 8);
  f.q[0] = ((const v4u*)p)[0];
  f.q[1] = ((const v4u*)p)[1];
}

// ---------------- weight repack: f32 -> bf16, column-major pair-packed [n][k2] -------
__global__ __launch_bounds__(256) void pack_weight(const float* __restrict__ src,
                                                   u32* __restrict__ dst,
                                                   int K, int N, int qkv) {
  int idx = blockIdx.x * 256 + threadIdx.x;
  int K2 = K >> 1;
  if (idx >= K2 * N) return;
  int k2 = idx % K2, n = idx / K2;             // k2 fast -> coalesced writes
  float lo, hi;
  if (qkv) {                                   // src (H, K, 128): W[k][h*128+d]=src[h][k][d]
    int h = n >> 7, d = n & 127;
    size_t base = (size_t)h * K * 128 + d;
    lo = src[base + (size_t)(2 * k2) * 128];
    hi = src[base + (size_t)(2 * k2 + 1) * 128];
  } else {
    lo = src[(size_t)(2 * k2) * N + n];
    hi = src[(size_t)(2 * k2 + 1) * N + n];
  }
  dst[(size_t)n * K2 + k2] = pk2bf(lo, hi);
}

// ---------------- LayerNorm (C=1024) -> bf16 ----------------
__global__ __launch_bounds__(256) void layernorm_bf16(const float* __restrict__ x,
                                                      const float* __restrict__ g,
                                                      const float* __restrict__ be,
                                                      u16* __restrict__ out) {
  const int C = 1024;
  __shared__ float red[8];
  int tid = threadIdx.x, lane = tid & 31, wave = tid >> 5;
  size_t base = (size_t)blockIdx.x * C;
  float v[4]; float s = 0.f;
#pragma unroll
  for (int i = 0; i < 4; i++) { v[i] = x[base + tid + i * 256]; s += v[i]; }
#pragma unroll
  for (int off = 16; off; off >>= 1) s += __shfl_xor(s, off, 32);
  if (lane == 0) red[wave] = s;
  __syncthreads();
  float tot = 0.f;
#pragma unroll
  for (int w = 0; w < 8; w++) tot += red[w];
  float mu = tot * (1.0f / 1024.0f);
  __syncthreads();
  float s2 = 0.f;
#pragma unroll
  for (int i = 0; i < 4; i++) { float d = v[i] - mu; s2 += d * d; }
#pragma unroll
  for (int off = 16; off; off >>= 1) s2 += __shfl_xor(s2, off, 32);
  if (lane == 0) red[wave] = s2;
  __syncthreads();
  float tot2 = 0.f;
#pragma unroll
  for (int w = 0; w < 8; w++) tot2 += red[w];
  float inv = rsqrtf(tot2 * (1.0f / 1024.0f) + 1e-5f);
#pragma unroll
  for (int i = 0; i < 4; i++) {
    int c = tid + i * 256;
    out[base + c] = f2bf((v[i] - mu) * inv * g[c] + be[c]);
  }
}

// ---------------- async double-buffered WMMA GEMM ----------------
// out = act(A(bf16 MxK row-major) * Bt(col-major pair-packed [n][K/2]) + bias [+resid])
// 128x128 block tile, BK=64, 8 waves (4x2), wave tile 32x64, 16 WMMA per stage.
enum { OUT_F32 = 0, OUT_BF16 = 1, OUT_PAIR = 2 };

#define GA_STRIDE 72                      // u16; 144B rows
#define GA_BYTES  (128 * GA_STRIDE * 2)   // 18432 per stage
#define GB_STRIDE 36                      // dwords; 144B rows [n][k2]
#define GB_BYTES  (128 * GB_STRIDE * 4)   // 18432 per stage
#define GEMM_LDS  (2 * GA_BYTES + 2 * GB_BYTES)   // 73728

__device__ __forceinline__ void gemm_issue(u64 abase, int ldA, u64 bbase, int ldBt,
                                           int m0, int n0, int k0,
                                           u32 ldsA, u32 ldsB, int tid) {
#pragma unroll
  for (int i = 0; i < 4; i++) {                // A: 128 rows x 8 chunks of 16B
    int idx = tid + i * 256;
    int r = idx >> 3, c = idx & 7;
    u32 voff = (u32)(m0 + r) * (u32)(ldA * 2) + (u32)(k0 * 2 + c * 16);
    async_b128(ldsA + r * (GA_STRIDE * 2) + c * 16, voff, abase);
  }
#pragma unroll
  for (int i = 0; i < 4; i++) {                // B: 128 n-rows x 8 chunks of 16B (32 k2)
    int idx = tid + i * 256;
    int n = idx >> 3, c = idx & 7;
    u32 voff = (u32)(n0 + n) * (u32)(ldBt * 4) + (u32)((k0 >> 1) * 4 + c * 16);
    async_b128(ldsB + n * (GB_STRIDE * 4) + c * 16, voff, bbase);
  }
}

template <int OUTMODE, bool RELU>
__global__ __launch_bounds__(256) void gemm_bf16(const u16* __restrict__ A, int ldA,
                                                 const u32* __restrict__ Bt, int ldBt,
                                                 int M, int N, int K,
                                                 const float* __restrict__ bias,
                                                 const float* __restrict__ resid,
                                                 void* __restrict__ out) {
  extern __shared__ char smem[];
  u32 lds0 = lds_addr(smem);
  u64 abase = (u64)(uintptr_t)A, bbase = (u64)(uintptr_t)Bt;
  int tid = threadIdx.x, lane = tid & 31, wave = tid >> 5;
  int wm = wave & 3, wn = wave >> 2;
  int m0 = blockIdx.x * 128, n0 = blockIdx.y * 128;

  v8f acc[2][4];
#pragma unroll
  for (int mt = 0; mt < 2; mt++)
#pragma unroll
    for (int nt = 0; nt < 4; nt++) acc[mt][nt] = (v8f)(0.0f);

  const int KS = K >> 6;
  gemm_issue(abase, ldA, bbase, ldBt, m0, n0, 0, lds0, lds0 + 2 * GA_BYTES, tid);

  for (int ks = 0; ks < KS; ks++) {
    int cur = ks & 1;
    __syncthreads();                           // stage ks-1 consumers done -> buffer free
    if (ks + 1 < KS) {
      int nxt = (ks + 1) & 1;
      gemm_issue(abase, ldA, bbase, ldBt, m0, n0, (ks + 1) * 64,
                 lds0 + nxt * GA_BYTES, lds0 + 2 * GA_BYTES + nxt * GB_BYTES, tid);
      asm volatile("s_wait_asynccnt 0x8" ::: "memory");  // stage ks landed
    } else {
      asm volatile("s_wait_asynccnt 0x0" ::: "memory");
    }
    __syncthreads();                           // all waves' stage ks data visible

    const u16* pA = (const u16*)(smem + cur * GA_BYTES);
    const u32* pB = (const u32*)(smem + 2 * GA_BYTES + cur * GB_BYTES);
    // preload all 12 fragments, then 16 back-to-back WMMAs (only D->C chains)
    FragBF af[2][2], bq[2][4];
#pragma unroll
    for (int kk = 0; kk < 2; kk++) {
#pragma unroll
      for (int mt = 0; mt < 2; mt++)
        load_a_frag(af[kk][mt], pA + (wm * 32 + mt * 16) * GA_STRIDE + kk * 32,
                    GA_STRIDE, lane);
#pragma unroll
      for (int nt = 0; nt < 4; nt++)
        load_b_frag_t(bq[kk][nt], pB + (wn * 64 + nt * 16) * GB_STRIDE + kk * 16,
                      GB_STRIDE, lane);
    }
#pragma unroll
    for (int mt = 0; mt < 2; mt++)
#pragma unroll
      for (int nt = 0; nt < 4; nt++)
#pragma unroll
        for (int kk = 0; kk < 2; kk++)
          acc[mt][nt] = wmma_bf16(af[kk][mt], bq[kk][nt], acc[mt][nt]);
  }

  int n_base = n0 + wn * 64 + (lane & 15);
  int m_base = m0 + wm * 32 + ((lane < 16) ? 0 : 8);
#pragma unroll
  for (int mt = 0; mt < 2; mt++) {
#pragma unroll
    for (int nt = 0; nt < 4; nt++) {
      int nn = n_base + nt * 16;
      if constexpr (OUTMODE == OUT_PAIR) {
        // v output, transposed pair-packed: [(b*8+h)*128+d][t2], row len T/2=1024
        int hh = nn >> 7, d = nn & 127;
#pragma unroll
        for (int i = 0; i < 4; i++) {
          u32 pk = pk2bf(acc[mt][nt][2 * i], acc[mt][nt][2 * i + 1]);
          int mp = (m_base + mt * 16 + 2 * i) >> 1;   // global pair row
          int bb = mp >> 10, t2 = mp & 1023;          // T/2 = 1024
          ((u32*)out)[((size_t)((bb * 8 + hh) * 128 + d) << 10) + t2] = pk;
        }
      } else {
#pragma unroll
        for (int r = 0; r < 8; r++) {
          float val = acc[mt][nt][r];
          int mr = m_base + mt * 16 + r;
          if (bias) val += bias[nn];
          if constexpr (RELU) val = fmaxf(val, 0.f);
          if (resid) val += resid[(size_t)mr * N + nn];
          if constexpr (OUTMODE == OUT_F32)
            ((float*)out)[(size_t)mr * N + nn] = val;
          else
            ((u16*)out)[(size_t)mr * N + nn] = f2bf(val);
        }
      }
    }
  }
}

// ---------------- flash attention: xa = x + softmax(q k^T * sqrt(Dh)) v ----------------
#define AQ_STRIDE 136                      // u16; 272B rows
#define AQ_BYTES  (128 * AQ_STRIDE * 2)    // 34816
#define AV_STRIDE 68                       // dwords; 272B rows [d][k2]
#define AV_BYTES  (128 * AV_STRIDE * 4)    // 34816
#define ATTN_LDS  (2 * AQ_BYTES + AV_BYTES)   // 104448

__global__ __launch_bounds__(256) void attention_kernel(const u16* __restrict__ q,
                                                        const u16* __restrict__ k,
                                                        const u32* __restrict__ vpt,
                                                        const float* __restrict__ x,
                                                        float* __restrict__ xa) {
  const int T = 2048, HD = 1024, Dh = 128;
  extern __shared__ char smem[];
  u16* sQ = (u16*)smem;                        // 128x136 bf16 (reused as P scratch)
  u16* sK = (u16*)(smem + AQ_BYTES);
  u32* sV = (u32*)(smem + 2 * AQ_BYTES);       // transposed [d][k2]
  u32 ldsQ = lds_addr(smem), ldsK = ldsQ + AQ_BYTES, ldsV = ldsQ + 2 * AQ_BYTES;
  int tid = threadIdx.x, lane = tid & 31, wave = tid >> 5;
  int bh = blockIdx.y, b = bh >> 3, h = bh & 7;
  int t0 = blockIdx.x * 128;
  u64 qbase = (u64)(uintptr_t)(q + (size_t)b * T * HD + (size_t)h * Dh);
  u64 kbase = (u64)(uintptr_t)(k + (size_t)b * T * HD + (size_t)h * Dh);
  u64 vbase = (u64)(uintptr_t)(vpt + ((size_t)bh * 128 << 10));

  // Q tile: 128 rows x 16 chunks of 16B (async)
#pragma unroll
  for (int i = 0; i < 8; i++) {
    int idx = tid + i * 256;
    int r = idx >> 4, c = idx & 15;
    async_b128(ldsQ + r * (AQ_STRIDE * 2) + c * 16,
               (u32)(t0 + r) * (u32)(HD * 2) + (u32)(c * 16), qbase);
  }
  asm volatile("s_wait_asynccnt 0x0" ::: "memory");
  __syncthreads();
  FragBF qf[4];
#pragma unroll
  for (int kk = 0; kk < 4; kk++)
    load_a_frag(qf[kk], sQ + (wave * 16) * AQ_STRIDE + kk * 32, AQ_STRIDE, lane);

  v8f o[8];
#pragma unroll
  for (int nt = 0; nt < 8; nt++) o[nt] = (v8f)(0.0f);
  float mrun[8], lrun[8];
#pragma unroll
  for (int r = 0; r < 8; r++) { mrun[r] = -1e30f; lrun[r] = 0.f; }
  const float scale = 11.3137084990f;          // sqrt(128): reference MULTIPLIES
  const float LOG2E = 1.4426950408889634f;
  u16* sP = sQ + (size_t)wave * 16 * AQ_STRIDE;   // private per-wave P slice (Q dead)

  for (int s0 = 0; s0 < T; s0 += 128) {
    __syncthreads();                           // prev sK/sV consumers done
#pragma unroll
    for (int i = 0; i < 8; i++) {              // K tile: 128 rows x 16 chunks
      int idx = tid + i * 256;
      int r = idx >> 4, c = idx & 15;
      async_b128(ldsK + r * (AQ_STRIDE * 2) + c * 16,
                 (u32)(s0 + r) * (u32)(HD * 2) + (u32)(c * 16), kbase);
    }
#pragma unroll
    for (int i = 0; i < 8; i++) {              // V tile: 128 d-rows x 16 chunks (64 k2)
      int idx = tid + i * 256;
      int d = idx >> 4, c = idx & 15;
      async_b128(ldsV + d * (AV_STRIDE * 4) + c * 16,
                 (u32)(d * 4096) + (u32)((s0 >> 1) * 4 + c * 16), vbase);
    }
    asm volatile("s_wait_asynccnt 0x0" ::: "memory");
    __syncthreads();

    // scores: double-buffered K^T fragments so WMMA sources are never load targets
    v8f st[8];
#pragma unroll
    for (int nt = 0; nt < 8; nt++) st[nt] = (v8f)(0.0f);
#pragma unroll
    for (int kk = 0; kk < 4; kk++) {
      FragBF kf[2];
      load_b_frag_rm(kf[0], sK + 0 * AQ_STRIDE + kk * 32, AQ_STRIDE, lane);
#pragma unroll
      for (int nt = 0; nt < 8; nt++) {
        if (nt < 7)
          load_b_frag_rm(kf[(nt + 1) & 1], sK + ((nt + 1) * 16) * AQ_STRIDE + kk * 32,
                         AQ_STRIDE, lane);
        st[nt] = wmma_bf16(qf[kk], kf[nt & 1], st[nt]);
      }
    }

    // online softmax (reductions stay within the 16-lane C-layout half)
#pragma unroll
    for (int r = 0; r < 8; r++) {
      float mx = -1e30f;
#pragma unroll
      for (int nt = 0; nt < 8; nt++) { st[nt][r] *= scale; mx = fmaxf(mx, st[nt][r]); }
#pragma unroll
      for (int off = 8; off; off >>= 1) mx = fmaxf(mx, __shfl_xor(mx, off, 32));
      float mnew = fmaxf(mrun[r], mx);
      float alpha = exp2f((mrun[r] - mnew) * LOG2E);
      mrun[r] = mnew;
      lrun[r] *= alpha;
#pragma unroll
      for (int nt = 0; nt < 8; nt++) o[nt][r] *= alpha;
      float rs = 0.f;
#pragma unroll
      for (int nt = 0; nt < 8; nt++) {
        float pv = exp2f((st[nt][r] - mnew) * LOG2E);
        st[nt][r] = pv; rs += pv;
      }
#pragma unroll
      for (int off = 8; off; off >>= 1) rs += __shfl_xor(rs, off, 32);
      lrun[r] += rs;
    }

    // C-layout -> A-layout via private LDS round trip
    {
      int prow = (lane < 16) ? 0 : 8;
      int pcol = lane & 15;
#pragma unroll
      for (int nt = 0; nt < 8; nt++)
#pragma unroll
        for (int r = 0; r < 8; r++)
          sP[(prow + r) * AQ_STRIDE + nt * 16 + pcol] = f2bf(st[nt][r]);
    }
    asm volatile("s_wait_dscnt 0" ::: "memory");   // P writes -> P reads, same wave

    // O += P @ V, double-buffered V fragments
#pragma unroll
    for (int kk = 0; kk < 4; kk++) {
      FragBF pf;
      load_a_frag(pf, sP + kk * 32, AQ_STRIDE, lane);
      FragBF vf[2];
      load_b_frag_t(vf[0], sV + 0 * AV_STRIDE + kk * 16, AV_STRIDE, lane);
#pragma unroll
      for (int nt = 0; nt < 8; nt++) {
        if (nt < 7)
          load_b_frag_t(vf[(nt + 1) & 1], sV + ((nt + 1) * 16) * AV_STRIDE + kk * 16,
                        AV_STRIDE, lane);
        o[nt] = wmma_bf16(pf, vf[nt & 1], o[nt]);
      }
    }
  }

  float invl[8];
#pragma unroll
  for (int r = 0; r < 8; r++) invl[r] = 1.0f / lrun[r];
  int half8 = (lane < 16) ? 0 : 8;
#pragma unroll
  for (int nt = 0; nt < 8; nt++) {
#pragma unroll
    for (int r = 0; r < 8; r++) {
      int trow = t0 + wave * 16 + half8 + r;
      int col = h * Dh + nt * 16 + (lane & 15);
      size_t gi = ((size_t)b * T + trow) * HD + col;
      xa[gi] = x[gi] + o[nt][r] * invl[r];
    }
  }
}

// ---------------- host ----------------

extern "C" void kernel_launch(void* const* d_in, const int* in_sizes, int n_in,
                              void* d_out, int out_size, void* d_ws, size_t ws_size,
                              hipStream_t stream) {
  const float* x     = (const float*)d_in[0];
  const float* wq    = (const float*)d_in[1];
  const float* wk    = (const float*)d_in[2];
  const float* wv    = (const float*)d_in[3];
  const float* w1    = (const float*)d_in[4];
  const float* b1    = (const float*)d_in[5];
  const float* w2    = (const float*)d_in[6];
  const float* b2    = (const float*)d_in[7];
  const float* ln1_g = (const float*)d_in[8];
  const float* ln1_b = (const float*)d_in[9];
  const float* ln2_g = (const float*)d_in[10];
  const float* ln2_b = (const float*)d_in[11];

  const int B = 4, T = 2048, C = 1024, Dff = 4096;
  const int M = B * T;                           // 8192

  char* p = (char*)d_ws;
  auto alloc = [&](size_t bytes) -> void* {
    void* r = (void*)p;
    p += (bytes + 255) & ~(size_t)255;
    return r;
  };
  u16* h    = (u16*)alloc((size_t)M * C * 2);
  u32* wqp  = (u32*)alloc((size_t)(C / 2) * C * 4);
  u32* wkp  = (u32*)alloc((size_t)(C / 2) * C * 4);
  u32* wvp  = (u32*)alloc((size_t)(C / 2) * C * 4);
  u32* w1p  = (u32*)alloc((size_t)(C / 2) * Dff * 4);
  u32* w2p  = (u32*)alloc((size_t)(Dff / 2) * C * 4);
  u16* qbuf = (u16*)alloc((size_t)M * C * 2);
  u16* kbuf = (u16*)alloc((size_t)M * C * 2);
  u32* vpt  = (u32*)alloc((size_t)(M / 2) * C * 4);
  float* xa = (float*)alloc((size_t)M * C * 4);
  u16* h2   = (u16*)alloc((size_t)M * C * 2);
  u16* ff1  = (u16*)alloc((size_t)M * Dff * 2);

  pack_weight<<<(512 * 1024 + 255) / 256, 256, 0, stream>>>(wq, wqp, C, C, 1);
  pack_weight<<<(512 * 1024 + 255) / 256, 256, 0, stream>>>(wk, wkp, C, C, 1);
  pack_weight<<<(512 * 1024 + 255) / 256, 256, 0, stream>>>(wv, wvp, C, C, 1);
  pack_weight<<<(512 * 4096 + 255) / 256, 256, 0, stream>>>(w1, w1p, C, Dff, 0);
  pack_weight<<<(2048 * 1024 + 255) / 256, 256, 0, stream>>>(w2, w2p, Dff, C, 0);

  layernorm_bf16<<<M, 256, 0, stream>>>(x, ln1_g, ln1_b, h);

  gemm_bf16<OUT_BF16, false><<<dim3(M / 128, C / 128), 256, GEMM_LDS, stream>>>(
      h, C, wqp, C / 2, M, C, C, nullptr, nullptr, qbuf);
  gemm_bf16<OUT_BF16, false><<<dim3(M / 128, C / 128), 256, GEMM_LDS, stream>>>(
      h, C, wkp, C / 2, M, C, C, nullptr, nullptr, kbuf);
  gemm_bf16<OUT_PAIR, false><<<dim3(M / 128, C / 128), 256, GEMM_LDS, stream>>>(
      h, C, wvp, C / 2, M, C, C, nullptr, nullptr, vpt);

  attention_kernel<<<dim3(T / 128, B * 8), 256, ATTN_LDS, stream>>>(qbuf, kbuf, vpt, x, xa);

  layernorm_bf16<<<M, 256, 0, stream>>>(xa, ln2_g, ln2_b, h2);

  gemm_bf16<OUT_BF16, true><<<dim3(M / 128, Dff / 128), 256, GEMM_LDS, stream>>>(
      h2, C, w1p, Dff / 2, M, Dff, C, b1, nullptr, ff1);
  gemm_bf16<OUT_F32, false><<<dim3(M / 128, C / 128), 256, GEMM_LDS, stream>>>(
      ff1, Dff, w2p, C / 2, M, C, Dff, b2, xa, (float*)d_out);
}
